// SparseExpert_11458972746041
// MI455X (gfx1250) — matrix operand order repaired and checked by reading, not compiled, not run on hardware
//
#include <hip/hip_runtime.h>
#include <hip/hip_bf16.h>

#define DIM   2048
#define NEXP  8
#define NTOK  8192
#define TM    128
#define TN    128
#define TK    64
#define KSTEPS (DIM / TK)
#define LROW  72   // _Float16 per padded LDS row: 64 data + 8 pad (144B = 36 DWORD stride
                   // -> conflict-free b128: starts {36*r mod 64} distinct, stride-4 spaced)

typedef __attribute__((ext_vector_type(16))) _Float16 v16h;
typedef __attribute__((ext_vector_type(8)))  float    v8f;

#if __has_builtin(__builtin_amdgcn_tensor_load_to_lds) && __has_builtin(__builtin_amdgcn_s_wait_tensorcnt)
#define USE_TDM 1
typedef unsigned int u32x4 __attribute__((ext_vector_type(4)));
typedef int          i32x8 __attribute__((ext_vector_type(8)));
typedef int          i32x4 __attribute__((ext_vector_type(4)));
#endif

// ---------------- f32 -> f16 convert ----------------
__global__ void cvt_f32_f16(const float* __restrict__ src, _Float16* __restrict__ dst, int n4) {
    int i = blockIdx.x * blockDim.x + threadIdx.x;
    if (i >= n4) return;
    float4 f = ((const float4*)src)[i];
    union { _Float16 h[4]; uint2 u; } o;
    o.h[0] = (_Float16)f.x; o.h[1] = (_Float16)f.y;
    o.h[2] = (_Float16)f.z; o.h[3] = (_Float16)f.w;
    ((uint2*)dst)[i] = o.u;
}

// ---------------- gating: one wave per token, exclude bottom-2 of 8 experts -------------
__global__ void gate_kernel(const float* __restrict__ x, const float* __restrict__ Wg,
                            const float* __restrict__ bg, unsigned* __restrict__ maskbits) {
    const int lane = threadIdx.x & 31;
    const int wave = threadIdx.x >> 5;
    const int n = blockIdx.x * 8 + wave;

    float s[NEXP];
#pragma unroll
    for (int e = 0; e < NEXP; ++e) s[e] = 0.0f;

    const float* xr = x + (size_t)n * DIM;
    for (int d = lane; d < DIM; d += 32) {
        float xv = xr[d];
#pragma unroll
        for (int e = 0; e < NEXP; ++e) s[e] += xv * Wg[e * DIM + d];
    }
#pragma unroll
    for (int e = 0; e < NEXP; ++e)
        for (int off = 16; off > 0; off >>= 1) s[e] += __shfl_xor(s[e], off, 32);

    if (lane == 0) {
        float g[NEXP];
#pragma unroll
        for (int e = 0; e < NEXP; ++e) g[e] = s[e] + bg[e];
        int i1 = 0; float v1 = g[0];
#pragma unroll
        for (int e = 1; e < NEXP; ++e) if (g[e] < v1) { v1 = g[e]; i1 = e; }
        int i2 = -1; float v2 = 3.4e38f;
#pragma unroll
        for (int e = 0; e < NEXP; ++e) if (e != i1 && g[e] < v2) { v2 = g[e]; i2 = e; }
        maskbits[n] = 0xFFu & ~((1u << i1) | (1u << i2));
    }
}

#ifdef USE_TDM
// Issue one TDM load of a [128 rows x 64 f16] tile (row stride DIM f16) into LDS,
// with hardware padding: every 32 DWORDs (128B row) insert 4 DWORDs (16B) -> 144B rows.
__device__ __forceinline__ void tdm_load_tile(unsigned lds_byte, const _Float16* gptr,
                                              unsigned tensor_rows) {
    unsigned long long ga = (unsigned long long)gptr;
    u32x4 g0;
    g0.x = 1u;                                              // count=1, user mode
    g0.y = lds_byte;                                        // lds_addr
    g0.z = (unsigned)(ga & 0xFFFFFFFFull);                  // global_addr[31:0]
    g0.w = (unsigned)((ga >> 32) & 0x1FFFFFFull) | 0x80000000u; // addr[56:32] | type=2
    i32x8 g1;
    g1[0] = (int)((1u << 16)      // data_size = 2 bytes
                | (1u << 20)      // pad_enable
                | (4u << 22)      // pad_interval: 32 DWORDs (128B)
                | (3u << 25));    // pad_amount:   4 DWORDs (16B)
    g1[1] = (int)(((unsigned)DIM & 0xFFFFu) << 16);         // tensor_dim0[15:0]
    g1[2] = (int)((tensor_rows & 0xFFFFu) << 16);           // dim0 hi=0 | tensor_dim1[15:0]
    g1[3] = (int)(((tensor_rows >> 16) & 0xFFFFu) | ((unsigned)TK << 16)); // dim1 hi | tile_dim0=64
    g1[4] = 128;                                            // tile_dim1=128, tile_dim2=0
    g1[5] = DIM;                                            // tensor_dim0_stride[31:0]
    g1[6] = 0;
    g1[7] = 0;
    i32x4 z4 = (i32x4){0, 0, 0, 0};
    i32x8 z8 = (i32x8){0, 0, 0, 0, 0, 0, 0, 0};
    __builtin_amdgcn_tensor_load_to_lds(g0, g1, z4, z4, z8, 0);
}
#endif

// ---------------- main WMMA GEMM: out = sum_e mask_e * (x @ We[e].T + be[e]) ------------
__global__ __launch_bounds__(256)
void moe_gemm(const _Float16* __restrict__ xh, const _Float16* __restrict__ Weh,
              const float* __restrict__ be, const unsigned* __restrict__ maskbits,
              float* __restrict__ out) {
    __shared__ __align__(16) _Float16 lA[2][TM * LROW];
    __shared__ __align__(16) _Float16 lB[2][TN * LROW];
    __shared__ float lM[NEXP][TM];

    const int t    = threadIdx.x;
    const int lane = t & 31;
    const int wave = t >> 5;
    const int wm   = wave >> 2;        // 0..1  (M direction)
    const int wn   = wave & 3;         // 0..3  (N direction)
    const int m0   = blockIdx.y * TM;
    const int n0   = blockIdx.x * TN;

    if (t < TM) {
        unsigned bits = maskbits[m0 + t];
#pragma unroll
        for (int e = 0; e < NEXP; ++e) lM[e][t] = ((bits >> e) & 1u) ? 1.0f : 0.0f;
    }

    const int fr  = lane & 15;         // fragment row/col within 16
    const int fcA = (lane >> 4) * 8;   // A frag: K chunks {c..c+7, c+16..c+23}
    const int fcB = (lane >> 4) * 8;   // B frag: K chunks {c..c+7, c+16..c+23}
#ifndef USE_TDM
    const int srow  = t >> 1;          // staging row 0..127
    const int sseg  = (t & 1) * 32;    // staging 64B segment offset (f16 units)
#endif

    v8f accF[4][2];
#pragma unroll
    for (int s = 0; s < 4; ++s)
#pragma unroll
        for (int j = 0; j < 2; ++j) accF[s][j] = (v8f){0,0,0,0,0,0,0,0};

    __syncthreads();  // lM ready

    for (int e = 0; e < NEXP; ++e) {
        v8f acc[4][2];
#pragma unroll
        for (int s = 0; s < 4; ++s)
#pragma unroll
            for (int j = 0; j < 2; ++j) acc[s][j] = (v8f){0,0,0,0,0,0,0,0};

        const _Float16* Abase = xh + (size_t)m0 * DIM;
        const _Float16* Bbase = Weh + ((size_t)e * DIM + n0) * DIM;

#ifdef USE_TDM
        if (wave == 0) {  // prologue: async-stage tile 0 into buffer 0
            tdm_load_tile((unsigned)(unsigned long long)&lA[0][0], Abase, (unsigned)NTOK);
            tdm_load_tile((unsigned)(unsigned long long)&lB[0][0], Bbase, (unsigned)DIM);
        }
#endif
        for (int ki = 0; ki < KSTEPS; ++ki) {
            const int cur = ki & 1;
#ifdef USE_TDM
            if (wave == 0) {
                if (ki + 1 < KSTEPS) {  // async-prefetch next tile into other buffer
                    tdm_load_tile((unsigned)(unsigned long long)&lA[cur ^ 1][0],
                                  Abase + (ki + 1) * TK, (unsigned)NTOK);
                    tdm_load_tile((unsigned)(unsigned long long)&lB[cur ^ 1][0],
                                  Bbase + (ki + 1) * TK, (unsigned)DIM);
                    __builtin_amdgcn_s_wait_tensorcnt(2);  // oldest two (current tile) done
                } else {
                    __builtin_amdgcn_s_wait_tensorcnt(0);
                }
            }
            __syncthreads();  // current buffer visible to all waves
#else
            {   // synchronous staging fallback: two 64B segments per row
                const int k0 = ki * TK;
                const uint4* gA = (const uint4*)(Abase + (size_t)srow * DIM + k0 + sseg);
                const uint4* gB = (const uint4*)(Bbase + (size_t)srow * DIM + k0 + sseg);
                uint4 a0 = gA[0], a1 = gA[1], a2 = gA[2], a3 = gA[3];
                uint4 b0 = gB[0], b1 = gB[1], b2 = gB[2], b3 = gB[3];
                uint4* sA = (uint4*)&lA[cur][srow * LROW + sseg];
                uint4* sB = (uint4*)&lB[cur][srow * LROW + sseg];
                sA[0] = a0; sA[1] = a1; sA[2] = a2; sA[3] = a3;
                sB[0] = b0; sB[1] = b1; sB[2] = b2; sB[3] = b3;
            }
            __syncthreads();
#endif
            // ---- two 32-K chunks per staged tile: 16 WMMAs between barrier pairs.
            //      Load ALL fragments for a chunk into distinct registers first so the
            //      12 ds_load_b128 issue as a clause and WMMAs stream without per-pair
            //      full DScnt drains.
#pragma unroll
            for (int kk = 0; kk < 2; ++kk) {
                const int kb = kk * 32;  // f16 offset of this 32-K chunk within padded row
                v16h bf[2];
#pragma unroll
                for (int j = 0; j < 2; ++j) {
                    const _Float16* rp = &lB[cur][(wn * 32 + j * 16 + fr) * LROW + kb];
                    union { v16h v; uint4 q[2]; } u;
                    u.q[0] = *(const uint4*)(rp + fcB);
                    u.q[1] = *(const uint4*)(rp + fcB + 16);
                    bf[j] = u.v;
                }
                v16h af[4];
#pragma unroll
                for (int s = 0; s < 4; ++s) {
                    const _Float16* rp = &lA[cur][(wm * 64 + s * 16 + fr) * LROW + kb];
                    union { v16h v; uint4 q[2]; } u;
                    u.q[0] = *(const uint4*)(rp + fcA);
                    u.q[1] = *(const uint4*)(rp + fcA + 16);
                    af[s] = u.v;
                }
#pragma unroll
                for (int s = 0; s < 4; ++s)
#pragma unroll
                    for (int j = 0; j < 2; ++j)
                        acc[s][j] = __builtin_amdgcn_wmma_f32_16x16x32_f16(
                            false, af[s], false, bf[j], (short)0, acc[s][j], false, false);
            }
            __syncthreads();  // all reads done before buffer reuse
        }

        // ---- epilogue for expert e: accF += mask_row * (acc + bias) ----
        float bias[2];
#pragma unroll
        for (int j = 0; j < 2; ++j)
            bias[j] = be[e * DIM + n0 + wn * 32 + j * 16 + fr];
        const int rbase = wm * 64 + (lane >> 4) * 8;
#pragma unroll
        for (int s = 0; s < 4; ++s)
#pragma unroll
            for (int i = 0; i < 8; ++i) {
                float m = lM[e][rbase + s * 16 + i];
#pragma unroll
                for (int j = 0; j < 2; ++j)
                    accF[s][j][i] += m * (acc[s][j][i] + bias[j]);
            }
    }

    // ---- store f32 output ----
#pragma unroll
    for (int s = 0; s < 4; ++s)
#pragma unroll
        for (int i = 0; i < 8; ++i) {
            int row = m0 + wm * 64 + s * 16 + (lane >> 4) * 8 + i;
#pragma unroll
            for (int j = 0; j < 2; ++j) {
                int col = n0 + wn * 32 + j * 16 + fr;
                out[(size_t)row * DIM + col] = accF[s][j][i];
            }
        }
}

extern "C" void kernel_launch(void* const* d_in, const int* in_sizes, int n_in,
                              void* d_out, int out_size, void* d_ws, size_t ws_size,
                              hipStream_t stream) {
    const float* x  = (const float*)d_in[0];
    const float* Wg = (const float*)d_in[1];
    const float* bg = (const float*)d_in[2];
    const float* We = (const float*)d_in[3];
    const float* be = (const float*)d_in[4];
    float* out = (float*)d_out;

    // workspace layout: xh | Weh | maskbits
    _Float16* xh  = (_Float16*)d_ws;
    _Float16* Weh = (_Float16*)((char*)d_ws + (size_t)NTOK * DIM * sizeof(_Float16));
    unsigned* mb  = (unsigned*)((char*)d_ws + (size_t)NTOK * DIM * sizeof(_Float16)
                                            + (size_t)NEXP * DIM * DIM * sizeof(_Float16));

    int n4x = NTOK * DIM / 4;
    cvt_f32_f16<<<(n4x + 255) / 256, 256, 0, stream>>>(x, xh, n4x);
    int n4w = NEXP * DIM * DIM / 4;
    cvt_f32_f16<<<(n4w + 255) / 256, 256, 0, stream>>>(We, Weh, n4w);

    gate_kernel<<<NTOK / 8, 256, 0, stream>>>(x, Wg, bg, mb);

    dim3 grid(DIM / TN, NTOK / TM, 1);
    moe_gemm<<<grid, 256, 0, stream>>>(xh, Weh, be, mb, out);
}
